// LSTM_35923106464338
// MI455X (gfx1250) — matrix operand order repaired and checked by reading, not compiled, not run on hardware
//
#include <hip/hip_runtime.h>
#include <cmath>

typedef __attribute__((ext_vector_type(16))) __bf16 v16bf;
typedef __attribute__((ext_vector_type(8)))  __bf16 v8bf;
typedef __attribute__((ext_vector_type(8)))  float  v8f;

#define TSTEPS 1024
#define BATCH  32
#define DIN    512
#define DH     1024
#define HTOT   1536   // DIN + DH
#define NGATE  4096   // 4*DH
#define KT_N   48     // HTOT/32 K-tiles total
#define NWG    64     // workgroups; each owns DH/NWG = 16 h-columns
#define NTHR   256    // 8 wave32s: 4 gates x 2 K-halves
#define AROW   1544   // padded A row length (stride 3088B == 16 mod 256 -> bank-rotate)

__device__ __forceinline__ float sigmoidf_(float x) {
    return 1.0f / (1.0f + __expf(-x));
}

// Branch-free tanh: avoids libm's divergent range checks on the critical path.
__device__ __forceinline__ float tanhf_(float x) {
#if __has_builtin(__builtin_amdgcn_tanhf)
    return __builtin_amdgcn_tanhf(x);
#elif __has_builtin(__builtin_amdgcn_tanh_f32)
    return __builtin_amdgcn_tanh_f32(x);
#else
    // tanh(x) = 1 - 2/(exp(2x)+1); __expf saturates cleanly for |x| large.
    return 1.0f - 2.0f / (__expf(2.0f * x) + 1.0f);
#endif
}

// Pack f32 weights [HTOT, NGATE] row-major into bf16 WMMA B-fragment layout:
// Bp[((nt*48 + kt)*32 + lane)*16 + e] = W[(kt*32 + (lane&16) + e)*NGATE + nt*16 + (lane&15)]
__global__ void lstm_prep_weights(const float* __restrict__ W,
                                  __bf16* __restrict__ Bp, int total) {
    int idx = blockIdx.x * blockDim.x + threadIdx.x;
    if (idx >= total) return;
    int e   = idx & 15;
    int tmp = idx >> 4;
    int l   = tmp & 31; tmp >>= 5;
    int kt  = tmp % KT_N;
    int nt  = tmp / KT_N;
    int n   = (nt << 4) + (l & 15);
    int K   = kt * 32 + (l & 16) + e;
    Bp[idx] = (__bf16)W[(size_t)K * NGATE + n];
}

// Transpose + convert x [b,t,d] f32 -> xT [t,b,d] bf16 (contiguous 32KB per step).
__global__ void lstm_prep_x(const float* __restrict__ x,
                            __bf16* __restrict__ xT, int total) {
    int idx = blockIdx.x * blockDim.x + threadIdx.x;
    if (idx >= total) return;
    int t = idx >> 14;            // / (BATCH*DIN)
    int r = idx & 16383;
    int b = r >> 9;
    int d = r & 511;
    xT[idx] = (__bf16)x[(size_t)b * (TSTEPS * DIN) + (size_t)t * DIN + d];
}

// Broadcast h_init to all 32 batch rows (bf16) and zero the barrier counter.
__global__ void lstm_prep_state(const float* __restrict__ h_init,
                                __bf16* __restrict__ hbf,
                                unsigned int* __restrict__ cnt) {
    int idx = blockIdx.x * blockDim.x + threadIdx.x;
    if (idx == 0) *cnt = 0u;
    if (idx < BATCH * DH) {
        int j = idx & (DH - 1);
        hbf[idx] = (__bf16)h_init[j];
    }
}

__global__ __launch_bounds__(NTHR)
void lstm_persistent(const __bf16* __restrict__ xT,    // [1024,32,512] bf16
                     const float*  __restrict__ bias,  // [4096] f32
                     const __bf16* __restrict__ Bp,    // packed bf16 weights
                     __bf16*       __restrict__ hbf,   // [32,1024] bf16 recurrent h
                     unsigned int* __restrict__ cnt,   // grid barrier counter
                     float*        __restrict__ out)   // [32,1024,1024] f32
{
    // All of this WG's weights, resident in LDS for the whole scan:
    __shared__ __bf16 ldsW[4][KT_N][32 * 16];        // 196608 B
    // Per-step activation matrix A = [x_t | h_{t-1}], padded rows:
    __shared__ __bf16 Abuf[BATCH][AROW];             //  98816 B
    // Partial gate pre-activations (2 K-halves x 4 gates):
    __shared__ float  gbuf[8][32][17];               //  17408 B
                                                     // total 312832 B < 320KB

    const int tid   = threadIdx.x;
    const int lane  = tid & 31;
    const int wave  = tid >> 5;
    const int gate  = wave & 3;         // 0..3 = f,i,o,g
    const int khalf = wave >> 2;        // K-split: kt 0..23 vs 24..47
    const int wg    = blockIdx.x;
    const int laneN = lane & 15;
    const int hi    = (lane >> 4) & 1;  // lane half (K-half selector in A/B layouts)
    const int ktbeg = khalf * 24;

    const int colbase = wg * 16;                       // h-column base of this WG
    const int n_glob  = gate * DH + colbase + laneN;   // gate column of this lane

    // One-time copy of this WG's full weight slice (192KB) to LDS.
    {
        uint4* dst = (uint4*)&ldsW[0][0][0];
        for (int i = tid; i < 12288; i += NTHR) {      // 12288 uint4 = 196608 B
            const int blk = i >> 6;                    // (g, kt) 1KB block
            const int sub = i & 63;
            const int g   = blk / KT_N;
            const int kt  = blk % KT_N;
            const uint4* s = (const uint4*)
                (Bp + ((size_t)(g * (DH / 16) + wg) * KT_N + kt) * 512) + sub;
            dst[i] = *s;
        }
    }

    // accumulator init: bias folded into khalf 0's partial
    const float binit = (khalf == 0) ? bias[n_glob] : 0.0f;
    v8f init_splat;
    #pragma unroll
    for (int r = 0; r < 8; ++r) init_splat[r] = binit;

    const __bf16* ldsWg = &ldsW[gate][0][lane * 16];   // B frag: stride 512/K-tile

    // A addressing (contiguous k order, matching Bp: lanes>=16 hold K 16..31)
    const int achunk = hi * 16;
    const __bf16* arow0 = &Abuf[laneN][achunk];        // M-tile 0 (batch 0..15)
    const __bf16* arow1 = &Abuf[laneN + 16][achunk];   // M-tile 1 (batch 16..31)

    float creg[2];                  // cell state, fixed (b,col) ownership per thread
    creg[0] = 0.0f; creg[1] = 0.0f;

    __syncthreads();                // ldsW ready

    for (int t = 0; t < TSTEPS; ++t) {
        // Stage A = [x_t | h_{t-1}] into LDS (single global read per WG/step).
        {
            const uint4* xsrc = (const uint4*)(xT + (size_t)t * (BATCH * DIN));
            for (int i = tid; i < 2048; i += NTHR) {        // 32KB x-slice
                const int b = i >> 6, d8 = i & 63;
                *(uint4*)&Abuf[b][d8 * 8] = xsrc[i];
            }
            const uint4* hsrc = (const uint4*)hbf;
            for (int i = tid; i < 4096; i += NTHR) {        // 64KB h state
                const int b = i >> 7, d8 = i & 127;
                *(uint4*)&Abuf[b][DIN + d8 * 8] = hsrc[i];
            }
        }
        __syncthreads();

        // GEMM entirely out of LDS: 24 K-tiles per wave, 2 WMMAs per tile.
        v8f acc0 = init_splat;
        v8f acc1 = init_splat;
        #pragma unroll 4
        for (int q = 0; q < 24; ++q) {
            const int kt = ktbeg + q;
            v16bf bfrag = *(const v16bf*)(ldsWg + kt * 512);
            union { v16bf v; v8bf h[2]; } a0u, a1u;
            a0u.h[0] = *(const v8bf*)(arow0 + kt * 32);
            a0u.h[1] = *(const v8bf*)(arow0 + kt * 32 + 8);
            a1u.h[0] = *(const v8bf*)(arow1 + kt * 32);
            a1u.h[1] = *(const v8bf*)(arow1 + kt * 32 + 8);
            acc0 = __builtin_amdgcn_wmma_f32_16x16x32_bf16(
                       false, a0u.v, false, bfrag, (short)0, acc0, false, false);
            acc1 = __builtin_amdgcn_wmma_f32_16x16x32_bf16(
                       false, a1u.v, false, bfrag, (short)0, acc1, false, false);
        }

        // Scatter partial gate pre-activations (C layout: VGPR r -> M = r + hi*8)
        const int pl = (khalf << 2) | gate;
        const int Mb = hi * 8;
        #pragma unroll
        for (int r = 0; r < 8; ++r) gbuf[pl][Mb + r][laneN]      = acc0[r];
        #pragma unroll
        for (int r = 0; r < 8; ++r) gbuf[pl][16 + Mb + r][laneN] = acc1[r];
        __syncthreads();

        // Elementwise LSTM update: 32 batch x 16 cols = 512 elems, 2 per thread
        #pragma unroll
        for (int k = 0; k < 2; ++k) {
            const int idx = tid * 2 + k;
            const int b   = idx >> 4;
            const int col = idx & 15;
            const float f = gbuf[0][b][col] + gbuf[4][b][col];
            const float i = gbuf[1][b][col] + gbuf[5][b][col];
            const float o = gbuf[2][b][col] + gbuf[6][b][col];
            const float g = gbuf[3][b][col] + gbuf[7][b][col];
            const float cn = sigmoidf_(f) * creg[k] + sigmoidf_(i) * tanhf_(g);
            creg[k] = cn;
            const float h = sigmoidf_(o) * tanhf_(cn);
            out[(size_t)b * (TSTEPS * DH) + (size_t)t * DH + colbase + col] = h;
            hbf[b * DH + colbase + col] = (__bf16)h;
        }

        // Device-wide barrier before next step reads hbf
        __threadfence();
        __syncthreads();
        if (tid == 0) {
            __hip_atomic_fetch_add(cnt, 1u, __ATOMIC_RELEASE,
                                   __HIP_MEMORY_SCOPE_AGENT);
            const unsigned int target = (unsigned int)(NWG * (t + 1));
            while (__hip_atomic_load(cnt, __ATOMIC_ACQUIRE,
                                     __HIP_MEMORY_SCOPE_AGENT) < target) {
                __builtin_amdgcn_s_sleep(1);
            }
        }
        __syncthreads();
    }
}

extern "C" void kernel_launch(void* const* d_in, const int* in_sizes, int n_in,
                              void* d_out, int out_size, void* d_ws, size_t ws_size,
                              hipStream_t stream) {
    (void)in_sizes; (void)n_in; (void)out_size; (void)ws_size;

    const float* x      = (const float*)d_in[0];
    // d_in[1] = input_paddings (unused by reference)
    const float* W      = (const float*)d_in[2];
    const float* bias   = (const float*)d_in[3];
    const float* h_init = (const float*)d_in[4];
    float* out = (float*)d_out;

    // Workspace layout:
    //   [0, 256)              : barrier counter
    //   [256, +64KB)          : hbf (32x1024 bf16 recurrent state)
    //   [.., +12.58MB)        : Bp  (packed bf16 weights)
    //   [.., +33.55MB)        : xT  (x transposed+converted to bf16, [t,b,d])
    char* ws = (char*)d_ws;
    unsigned int* cnt = (unsigned int*)ws;
    __bf16* hbf = (__bf16*)(ws + 256);
    __bf16* Bp  = (__bf16*)(ws + 256 + (size_t)BATCH * DH * 2);
    __bf16* xT  = (__bf16*)(ws + 256 + (size_t)BATCH * DH * 2
                               + (size_t)HTOT * NGATE * 2);

    const int totalW = (NGATE / 16) * KT_N * 32 * 16;   // 6,291,456
    const int totalX = TSTEPS * BATCH * DIN;            // 16,777,216
    lstm_prep_weights<<<totalW / 256, 256, 0, stream>>>(W, Bp, totalW);
    lstm_prep_x<<<totalX / 256, 256, 0, stream>>>(x, xT, totalX);
    lstm_prep_state<<<(BATCH * DH) / 256, 256, 0, stream>>>(h_init, hbf, cnt);
    lstm_persistent<<<NWG, NTHR, 0, stream>>>(xT, bias, Bp, hbf, cnt, out);
}
